// CodeDecoderWrapper_65884798320663
// MI455X (gfx1250) — matrix operand order repaired and checked by reading, not compiled, not run on hardware
//
#include <hip/hip_runtime.h>
#include <math.h>

#define L_   12
#define H_   1024
#define NH_  16
#define HD_  64
#define NKV_ 8
#define I_   4096
#define V_   8192
#define S_   256
#define KVD_ (NKV_*HD_)   /* 512 */

typedef _Float16 v16h __attribute__((ext_vector_type(16)));
typedef float    v8f  __attribute__((ext_vector_type(8)));
typedef float    f4   __attribute__((ext_vector_type(4)));

__device__ __forceinline__ float wave_sum(float v) {
#pragma unroll
  for (int o = 16; o > 0; o >>= 1) v += __shfl_xor(v, o, 32);
  return v;
}
__device__ __forceinline__ float wave_max(float v) {
#pragma unroll
  for (int o = 16; o > 0; o >>= 1) v = fmaxf(v, __shfl_xor(v, o, 32));
  return v;
}

// ---------------- init: hidden copy + rope tables ----------------
__global__ void k_init(const float* __restrict__ emb, const float* __restrict__ clen,
                       const float* __restrict__ invf, float* __restrict__ hidden,
                       float* __restrict__ cosb, float* __restrict__ sinb) {
  int t = threadIdx.x;                       // 1024 threads
  hidden[t] = emb[t];
  if (t < HD_) {
    float f = invf[t & (HD_/2 - 1)] * clen[0];
    cosb[t] = cosf(f);
    sinb[t] = sinf(f);
  }
}

// ---------------- RMSNorm over H=1024 (one block, 1024 threads) ----------------
__global__ void k_rms(const float* __restrict__ x, const float* __restrict__ w,
                      float* __restrict__ y, float* __restrict__ y2) {
  int t = threadIdx.x;
  float v = x[t];
  float ss = wave_sum(v * v);
  __shared__ float red[32];
  __shared__ float stot;
  int wid = t >> 5, lid = t & 31;
  if (lid == 0) red[wid] = ss;
  __syncthreads();
  if (wid == 0) {
    float s2 = wave_sum(red[lid]);
    if (lid == 0) stot = s2;
  }
  __syncthreads();
  float scale = rsqrtf(stot * (1.0f / H_) + 1e-6f);
  float o = v * scale * w[t];
  y[t] = o;
  if (y2) y2[t] = o;
}

// ---------------- fused QKV GEMV: 2048 rows, cols=1024 ----------------
__global__ void k_qkv(const float* __restrict__ qw, const float* __restrict__ kw,
                      const float* __restrict__ vw, const float* __restrict__ x,
                      float* __restrict__ q, float* __restrict__ k, float* __restrict__ v) {
  int row = blockIdx.x, t = threadIdx.x;
  const float* W; float* out; int r;
  if (row < 1024)      { W = qw; out = q; r = row; }
  else if (row < 1536) { W = kw; out = k; r = row - 1024; }
  else                 { W = vw; out = v; r = row - 1536; }
  const f4* w4 = (const f4*)(W + (size_t)r * H_);
  const f4* x4 = (const f4*)x;
  f4 wv = __builtin_nontemporal_load(w4 + t);
  f4 xv = x4[t];
  float sum = wv[0]*xv[0] + wv[1]*xv[1] + wv[2]*xv[2] + wv[3]*xv[3];
  sum = wave_sum(sum);
  __shared__ float red[8];
  if ((t & 31) == 0) red[t >> 5] = sum;
  __syncthreads();
  if (t == 0) {
    float tot = red[0]+red[1]+red[2]+red[3]+red[4]+red[5]+red[6]+red[7];
    out[r] = tot;
  }
}

// ---------------- per-head RMS + RoPE (24 blocks x 64 thr: 16 q heads, 8 k heads) ----
__global__ void k_qkrope(float* __restrict__ q, float* __restrict__ k,
                         const float* __restrict__ qn, const float* __restrict__ kn,
                         const float* __restrict__ cosb, const float* __restrict__ sinb) {
  __shared__ float buf[HD_];
  __shared__ float red[2];
  int h = blockIdx.x, d = threadIdx.x;
  float* x; const float* w;
  if (h < NH_) { x = q + h * HD_;        w = qn; }
  else         { x = k + (h - NH_) * HD_; w = kn; }
  float val = x[d];
  float ss = wave_sum(val * val);
  if ((d & 31) == 0) red[d >> 5] = ss;
  __syncthreads();
  float scale = rsqrtf((red[0] + red[1]) * (1.0f / HD_) + 1e-6f);
  float nv = val * scale * w[d];
  buf[d] = nv;
  __syncthreads();
  float rot = (d < HD_/2) ? -buf[d + HD_/2] : buf[d - HD_/2];
  x[d] = nv * cosb[d] + rot * sinb[d];
}

// ---------------- cache blend: out = cache*(1-um) + new*um (exact reference math) ----
__global__ void k_blend(const float* __restrict__ kcin, const float* __restrict__ vcin,
                        const float* __restrict__ knew, const float* __restrict__ vnew,
                        const float* __restrict__ um,
                        float* __restrict__ kcout, float* __restrict__ vcout) {
  int idx = blockIdx.x * 256 + threadIdx.x;      // 0 .. 2*KVD*S-1
  const int half = KVD_ * S_;
  if (idx < half) {
    int chan = idx >> 8, s = idx & (S_-1);
    float u = um[s];
    kcout[idx] = kcin[idx] * (1.0f - u) + knew[chan] * u;
  } else {
    int j = idx - half;
    int chan = j >> 8, s = j & (S_-1);
    float u = um[s];
    vcout[j] = vcin[j] * (1.0f - u) + vnew[chan] * u;
  }
}

// ---------------- attention via v_wmma_f32_16x16x32_f16 ----------------
// grid = 8 (kv heads), block = 128 (4 waves). Per kv head: 2 GQA query heads
// occupy rows 0,1 of the 16x16 WMMA tile; other rows zero.
__global__ void k_attn(const float* __restrict__ qbuf, const float* __restrict__ kc,
                       const float* __restrict__ vc, const float* __restrict__ mask,
                       float* __restrict__ ao) {
  int kvh = blockIdx.x;
  int tid = threadIdx.x;
  int wv = tid >> 5;
  int lane = tid & 31;
  __shared__ float aw[2][S_];

  int m  = lane & 15;
  int hi = (lane >= 16) ? 16 : 0;     // A-fragment K-half offset (contiguous k = 16*h + s, matched in B)
  float vmask = (m < 2) ? 1.0f : 0.0f;
  int mm = (m < 2) ? m : 0;
  int n  = lane & 15;                 // B/C/D column
  int kh = (lane >= 16) ? 16 : 0;     // B-fragment K-half offset

  // A fragments: q rows for the two GQA heads, K-chunks d=0..31 and d=32..63
  const float* qrow = qbuf + (2 * kvh + mm) * HD_;
  v16h a0, a1;
#pragma unroll
  for (int v = 0; v < 8; ++v) {
    int kl = hi + 2 * v;
    a0[2*v]   = (_Float16)(qrow[kl]          * vmask);
    a0[2*v+1] = (_Float16)(qrow[kl + 1]      * vmask);
    a1[2*v]   = (_Float16)(qrow[32 + kl]     * vmask);
    a1[2*v+1] = (_Float16)(qrow[32 + kl + 1] * vmask);
  }

  const float* krow = kc + (size_t)kvh * HD_ * S_;
  // QK^T: wave wv covers 4 column blocks of 16 positions
  for (int sb = wv * 4; sb < wv * 4 + 4; ++sb) {
    int s = sb * 16 + n;
    v8f c = {0.f,0.f,0.f,0.f,0.f,0.f,0.f,0.f};
    v16h b;
#pragma unroll
    for (int v = 0; v < 8; ++v) {
      int d = kh + 2 * v;
      b[2*v]   = (_Float16)krow[d * S_ + s];
      b[2*v+1] = (_Float16)krow[(d + 1) * S_ + s];
    }
    c = __builtin_amdgcn_wmma_f32_16x16x32_f16(false, a0, false, b, (short)0, c, false, false);
#pragma unroll
    for (int v = 0; v < 8; ++v) {
      int d = 32 + kh + 2 * v;
      b[2*v]   = (_Float16)krow[d * S_ + s];
      b[2*v+1] = (_Float16)krow[(d + 1) * S_ + s];
    }
    c = __builtin_amdgcn_wmma_f32_16x16x32_f16(false, a1, false, b, (short)0, c, false, false);
    if (lane < 16) {                       // D rows 0,1 live in VGPR 0,1 / lanes 0-15
      aw[0][s] = c[0] * 0.125f + mask[s];  // 1/sqrt(64)
      aw[1][s] = c[1] * 0.125f + mask[s];
    }
  }
  __syncthreads();

  // softmax (f32) — wave 0 handles head 0, wave 1 handles head 1
  if (wv < 2) {
    float vals[8];
    float mx = -1e30f;
#pragma unroll
    for (int i = 0; i < 8; ++i) { vals[i] = aw[wv][lane + 32*i]; mx = fmaxf(mx, vals[i]); }
    mx = wave_max(mx);
    float sum = 0.0f;
#pragma unroll
    for (int i = 0; i < 8; ++i) { vals[i] = expf(vals[i] - mx); sum += vals[i]; }
    sum = wave_sum(sum);
    float inv = 1.0f / sum;
#pragma unroll
    for (int i = 0; i < 8; ++i) aw[wv][lane + 32*i] = vals[i] * inv;
  }
  __syncthreads();

  // P·V: wave wv handles output dims [wv*16, wv*16+16)
  int dbase = wv * 16;
  const float* vrow = vc + (size_t)(kvh * HD_ + dbase + n) * S_;
  v8f c = {0.f,0.f,0.f,0.f,0.f,0.f,0.f,0.f};
  for (int kb = 0; kb < S_; kb += 32) {
    v16h a, b;
#pragma unroll
    for (int v = 0; v < 8; ++v) {
      int kl = hi + 2 * v;
      a[2*v]   = (_Float16)(aw[mm][kb + kl]     * vmask);
      a[2*v+1] = (_Float16)(aw[mm][kb + kl + 1] * vmask);
      int kk = kh + 2 * v;
      b[2*v]   = (_Float16)vrow[kb + kk];
      b[2*v+1] = (_Float16)vrow[kb + kk + 1];
    }
    c = __builtin_amdgcn_wmma_f32_16x16x32_f16(false, a, false, b, (short)0, c, false, false);
  }
  if (lane < 16) {
    ao[(2*kvh + 0) * HD_ + dbase + lane] = c[0];
    ao[(2*kvh + 1) * HD_ + dbase + lane] = c[1];
  }
}

// ---------------- generic GEMV: one block/row, 256 thr, cols multiple of 1024 ------
__global__ void k_gemv(const float* __restrict__ W, const float* __restrict__ x,
                       float* __restrict__ y, int cols, float beta) {
  int row = blockIdx.x, t = threadIdx.x;
  const f4* w4 = (const f4*)(W + (size_t)row * cols);
  const f4* x4 = (const f4*)x;
  float sum = 0.0f;
  int nIter = cols >> 10;                  // 256 thr * 4 floats = 1024/iter
  for (int i = 0; i < nIter; ++i) {
    int idx = i * 256 + t;
    f4 wv = __builtin_nontemporal_load(w4 + idx);
    f4 xv = x4[idx];
    sum += wv[0]*xv[0] + wv[1]*xv[1] + wv[2]*xv[2] + wv[3]*xv[3];
  }
  sum = wave_sum(sum);
  __shared__ float red[8];
  if ((t & 31) == 0) red[t >> 5] = sum;
  __syncthreads();
  if (t == 0) {
    float tot = red[0]+red[1]+red[2]+red[3]+red[4]+red[5]+red[6]+red[7];
    y[row] = beta * y[row] + tot;
  }
}

// ---------------- fused gate/up GEMV + SiLU*up (4096 rows, cols=1024) --------------
__global__ void k_gateup(const float* __restrict__ gw, const float* __restrict__ uw,
                         const float* __restrict__ x, float* __restrict__ act) {
  int row = blockIdx.x, t = threadIdx.x;
  const f4* g4 = (const f4*)(gw + (size_t)row * H_);
  const f4* u4 = (const f4*)(uw + (size_t)row * H_);
  const f4* x4 = (const f4*)x;
  f4 gv = __builtin_nontemporal_load(g4 + t);
  f4 uv = __builtin_nontemporal_load(u4 + t);
  f4 xv = x4[t];
  float gs = gv[0]*xv[0] + gv[1]*xv[1] + gv[2]*xv[2] + gv[3]*xv[3];
  float us = uv[0]*xv[0] + uv[1]*xv[1] + uv[2]*xv[2] + uv[3]*xv[3];
  gs = wave_sum(gs);
  us = wave_sum(us);
  __shared__ float redg[8], redu[8];
  if ((t & 31) == 0) { redg[t >> 5] = gs; redu[t >> 5] = us; }
  __syncthreads();
  if (t == 0) {
    float g = redg[0]+redg[1]+redg[2]+redg[3]+redg[4]+redg[5]+redg[6]+redg[7];
    float u = redu[0]+redu[1]+redu[2]+redu[3]+redu[4]+redu[5]+redu[6]+redu[7];
    act[row] = (g / (1.0f + expf(-g))) * u;   // silu(g) * u
  }
}

extern "C" void kernel_launch(void* const* d_in, const int* in_sizes, int n_in,
                              void* d_out, int out_size, void* d_ws, size_t ws_size,
                              hipStream_t stream) {
  (void)in_sizes; (void)n_in; (void)out_size; (void)ws_size;
  const float* input_embeds = (const float*)d_in[0];
  const float* cache_length = (const float*)d_in[1];
  const float* kp_mask      = (const float*)d_in[2];
  const float* um           = (const float*)d_in[3];
  const float* key_cache    = (const float*)d_in[4];
  const float* value_cache  = (const float*)d_in[5];
  const float* inv_freq     = (const float*)d_in[6];
  const float* ln1_w        = (const float*)d_in[7];
  const float* q_w          = (const float*)d_in[8];
  const float* k_w          = (const float*)d_in[9];
  const float* v_w          = (const float*)d_in[10];
  const float* qn_w         = (const float*)d_in[11];
  const float* kn_w         = (const float*)d_in[12];
  const float* o_w          = (const float*)d_in[13];
  const float* ln2_w        = (const float*)d_in[14];
  const float* gate_w       = (const float*)d_in[15];
  const float* up_w         = (const float*)d_in[16];
  const float* down_w       = (const float*)d_in[17];
  const float* norm_w       = (const float*)d_in[18];
  const float* head_w       = (const float*)d_in[19];

  float* out     = (float*)d_out;
  float* logits  = out;                         // V_
  float* hid_out = out + V_;                    // H_
  float* kco     = out + V_ + H_;               // L*KVD*S
  float* vco     = kco + (size_t)L_ * KVD_ * S_;

  float* ws     = (float*)d_ws;
  float* hidden = ws;            // 1024
  float* hn     = ws + 1024;     // 1024
  float* qb     = ws + 2048;     // 1024
  float* kb     = ws + 3072;     // 512
  float* vb     = ws + 3584;     // 512
  float* cosb   = ws + 4096;     // 64
  float* sinb   = ws + 4160;     // 64
  float* ao     = ws + 4224;     // 1024
  float* act    = ws + 5248;     // 4096

  k_init<<<1, 1024, 0, stream>>>(input_embeds, cache_length, inv_freq, hidden, cosb, sinb);

  for (int l = 0; l < L_; ++l) {
    const size_t qs = (size_t)l * NH_ * HD_ * H_;
    const size_t ks = (size_t)l * KVD_ * H_;
    const size_t cs = (size_t)l * KVD_ * S_;
    k_rms<<<1, 1024, 0, stream>>>(hidden, ln1_w + l * H_, hn, nullptr);
    k_qkv<<<2048, 256, 0, stream>>>(q_w + qs, k_w + ks, v_w + ks, hn, qb, kb, vb);
    k_qkrope<<<24, 64, 0, stream>>>(qb, kb, qn_w + l * HD_, kn_w + l * HD_, cosb, sinb);
    k_blend<<<(2 * KVD_ * S_) / 256, 256, 0, stream>>>(key_cache + cs, value_cache + cs,
                                                       kb, vb, um, kco + cs, vco + cs);
    k_attn<<<NKV_, 128, 0, stream>>>(qb, kco + cs, vco + cs, kp_mask, ao);
    k_gemv<<<H_, 256, 0, stream>>>(o_w + (size_t)l * H_ * NH_ * HD_, ao, hidden,
                                   NH_ * HD_, 1.0f);
    k_rms<<<1, 1024, 0, stream>>>(hidden, ln2_w + l * H_, hn, nullptr);
    k_gateup<<<I_, 256, 0, stream>>>(gate_w + (size_t)l * I_ * H_,
                                     up_w + (size_t)l * I_ * H_, hn, act);
    k_gemv<<<H_, 256, 0, stream>>>(down_w + (size_t)l * H_ * I_, act, hidden, I_, 1.0f);
  }

  k_rms<<<1, 1024, 0, stream>>>(hidden, norm_w, hn, hid_out);
  k_gemv<<<V_, 256, 0, stream>>>(head_w, hn, logits, H_, 0.0f);
}
